// Correlation_cal_31044023616046
// MI455X (gfx1250) — compile-verified, matrix operand order checked by reading, not run on hardware
//
#include <hip/hip_runtime.h>

// CDNA5 / gfx1250 fused kernel:
//   ref  = leaky_relu(conv1x1(ref_LR, w_first))          (VALU, from LDS)
//   att  = sigmoid(3x3 box-sum(hs * ref))                (VALU, bf16 LDS tiles)
//   out  = conv1x1([att*hs ; hs], w_fuse)                (v_wmma_f32_16x16x32_bf16)
//
// Tile: 32x2 output pixels per 256-thread block (8 wave32).
// Final GEMM per block: out[64o x 64px] = Wf[64x128] x act[128x64px] + b
//   -> 4 o-tiles x 4 px-tiles x 4 k-chunks of 16x16x32 bf16 WMMA.

typedef __attribute__((ext_vector_type(16))) __bf16 v16bf;
typedef __attribute__((ext_vector_type(8)))  float  v8f;

namespace {
constexpr int BN = 8, CN = 64, HN = 256, WN = 256;
constexpr int TPX = 32;            // tile width (pixels)
constexpr int TPY = 2;             // tile height (pixels)
constexpr int NPX = TPX * TPY;     // 64 pixels per block
constexpr int HALO_W = TPX + 2;    // 34
constexpr int HALO_H = TPY + 2;    // 4
constexpr int HPAD = 36;           // padded halo row (bf16 elems)
constexpr int KF = 2 * CN;         // 128 fused channels
constexpr int ALD = 144;           // act row stride (bf16) -> 288 B, 32B aligned
}

__device__ __forceinline__ int reflect_idx(int i, int n) {
    i = (i < 0) ? -i : i;
    i = (i >= n) ? (2 * n - 2 - i) : i;
    return i;
}

__global__ __launch_bounds__(256, 1)
void fused_corr_wmma(const float* __restrict__ hs,
                     const float* __restrict__ refLR,
                     const float* __restrict__ w_first,
                     const float* __restrict__ b_first,
                     const float* __restrict__ w_fuse,
                     const float* __restrict__ b_fuse,
                     float* __restrict__ out)
{
    __shared__ __bf16 s_hs [CN * HALO_H * HPAD];                      // 18432 B
    __shared__ __bf16 s_ref[CN * HALO_H * HPAD];                      // 18432 B
    __shared__ float  s_rlr[9 * HALO_H * HPAD];                       //  5184 B
    __shared__ __bf16 s_act[NPX * ALD] __attribute__((aligned(32)));  // 18432 B
    __shared__ float  s_wfirst[9 * CN];                               //  2304 B
    __shared__ float  s_bfirst[CN];
    __shared__ float  s_bfuse [CN];

    const int t = threadIdx.x;
    const int tilesX = WN / TPX;           // 8
    const int tilesY = HN / TPY;           // 128
    int bid = blockIdx.x;
    const int b  = bid / (tilesX * tilesY);
    int rem      = bid % (tilesX * tilesY);
    const int y0 = (rem / tilesX) * TPY;
    const int x0 = (rem % tilesX) * TPX;

    // ---- Phase 0: stage halos + weights into LDS ----------------------------
    for (int idx = t; idx < CN * HALO_H * HALO_W; idx += 256) {
        int cc = idx % HALO_W;
        int rr = (idx / HALO_W) % HALO_H;
        int c  = idx / (HALO_W * HALO_H);
        int yy = reflect_idx(y0 - 1 + rr, HN);
        int xx = reflect_idx(x0 - 1 + cc, WN);
        float v = hs[(((size_t)b * CN + c) * HN + yy) * WN + xx];
        s_hs[(c * HALO_H + rr) * HPAD + cc] = (__bf16)v;
    }
    for (int idx = t; idx < 9 * HALO_H * HALO_W; idx += 256) {
        int cc = idx % HALO_W;
        int rr = (idx / HALO_W) % HALO_H;
        int k  = idx / (HALO_W * HALO_H);
        int yy = reflect_idx(y0 - 1 + rr, HN);
        int xx = reflect_idx(x0 - 1 + cc, WN);
        s_rlr[(k * HALO_H + rr) * HPAD + cc] =
            refLR[(((size_t)b * 9 + k) * HN + yy) * WN + xx];
    }
    for (int idx = t; idx < CN * 9; idx += 256) {
        int c = idx / 9, k = idx % 9;
        s_wfirst[k * CN + c] = w_first[idx];       // store transposed [k][c]
    }
    if (t < CN) { s_bfirst[t] = b_first[t]; s_bfuse[t] = b_fuse[t]; }
    __syncthreads();

    // ---- Phase 1: ref = leaky_relu(conv1x1(ref_LR)) on the halo (bf16) ------
    for (int idx = t; idx < CN * HALO_H * HALO_W; idx += 256) {
        int cc = idx % HALO_W;
        int rr = (idx / HALO_W) % HALO_H;
        int c  = idx / (HALO_W * HALO_H);
        float acc = s_bfirst[c];
        #pragma unroll
        for (int k = 0; k < 9; ++k)
            acc = fmaf(s_wfirst[k * CN + c], s_rlr[(k * HALO_H + rr) * HPAD + cc], acc);
        acc = (acc > 0.0f) ? acc : 0.1f * acc;
        s_ref[(c * HALO_H + rr) * HPAD + cc] = (__bf16)acc;
    }
    __syncthreads();

    // ---- Phase 2: cross -> sigmoid -> fused activations act[px][k] (bf16) ---
    {
        const int px = t % NPX;        // 0..63
        const int cg = t / NPX;        // 0..3
        const int r  = px / TPX;       // 0..1
        const int j  = px % TPX;       // 0..31
        #pragma unroll
        for (int i = 0; i < 16; ++i) {
            int c = cg * 16 + i;
            const __bf16* hrow = &s_hs [(c * HALO_H + r) * HPAD + j];
            const __bf16* rrow = &s_ref[(c * HALO_H + r) * HPAD + j];
            float cross = 0.0f;
            #pragma unroll
            for (int dy = 0; dy < 3; ++dy)
                #pragma unroll
                for (int dx = 0; dx < 3; ++dx)
                    cross = fmaf((float)hrow[dy * HPAD + dx],
                                 (float)rrow[dy * HPAD + dx], cross);
            float att = 1.0f / (1.0f + __expf(-cross));
            float hc  = (float)hrow[HPAD + 1];                 // center hs value
            s_act[px * ALD + c]      = (__bf16)(att * hc);
            s_act[px * ALD + CN + c] = (__bf16)hc;
        }
    }
    __syncthreads();

    // ---- Phase 3: WMMA GEMM: out = Wf[64x128] x act[128x64px] + b_fuse ------
    {
        const int wave  = t / 32;
        const int lane  = t % 32;
        const int otile = wave & 3;                 // 4 o-tiles of 16 channels
        const int orow  = otile * 16 + (lane & 15); // A: M = lane%16
        const int khalf = (lane >> 4) * 16;         // A/B: K-half by lane/16

        // A fragments (w_fuse rows, f32 -> bf16), kept in VGPRs for the wave.
        v16bf afrag[4];
        #pragma unroll
        for (int kc = 0; kc < 4; ++kc) {
            #pragma unroll
            for (int i = 0; i < 16; ++i)
                afrag[kc][i] = (__bf16)w_fuse[orow * KF + kc * 32 + khalf + i];
        }

        #pragma unroll
        for (int p = 0; p < 2; ++p) {
            const int pxtile = (wave >> 2) * 2 + p;        // 4 px-tiles, 2/wave
            const int pxn = pxtile * 16 + (lane & 15);     // B: N = lane%16
            v8f acc = {};
            #pragma unroll
            for (int kc = 0; kc < 4; ++kc) {
                v16bf bfrag = *reinterpret_cast<const v16bf*>(
                    &s_act[pxn * ALD + kc * 32 + khalf]);
                acc = __builtin_amdgcn_wmma_f32_16x16x32_bf16(
                    /*neg_a=*/false, afrag[kc], /*neg_b=*/false, bfrag,
                    /*c_mod=*/(short)0, acc, /*reuse_a=*/false, /*reuse_b=*/false);
            }
            const int rpx = pxn / TPX;                     // row within tile
            const int jx  = pxn % TPX;                     // col within tile
            #pragma unroll
            for (int r = 0; r < 8; ++r) {
                int m = r + ((lane < 16) ? 0 : 8);         // D layout: M per lane half
                int o = otile * 16 + m;
                float v = acc[r] + s_bfuse[o];
                out[(((size_t)b * CN + o) * HN + (y0 + rpx)) * WN + (x0 + jx)] = v;
            }
        }
    }
}

extern "C" void kernel_launch(void* const* d_in, const int* in_sizes, int n_in,
                              void* d_out, int out_size, void* d_ws, size_t ws_size,
                              hipStream_t stream) {
    const float* hs      = (const float*)d_in[0];
    const float* refLR   = (const float*)d_in[1];
    const float* w_first = (const float*)d_in[2];
    const float* b_first = (const float*)d_in[3];
    const float* w_fuse  = (const float*)d_in[4];
    const float* b_fuse  = (const float*)d_in[5];
    float* out = (float*)d_out;

    const int nblocks = BN * (HN / TPY) * (WN / TPX);   // 8 * 128 * 8 = 8192
    fused_corr_wmma<<<dim3(nblocks), dim3(256), 0, stream>>>(
        hs, refLR, w_first, b_first, w_fuse, b_fuse, out);
}
